// EagerAttention_50251117363388
// MI455X (gfx1250) — compile-verified, matrix operand order checked by reading, not run on hardware
//
#include <hip/hip_runtime.h>
#include <stdint.h>

// ---------------------------------------------------------------------------
// EagerAttention for MI455X (gfx1250, wave32, WMMA)
//   K1: qkv = x @ Wqkv^T  -> q(bf16,scaled)[B,H,N,D], k(bf16)[B,H,N,D], v^T(bf16)[B,H,D,N]
//   K2: flash attention (online softmax); K/V tiles double-buffered in LDS via
//       global_load_async_to_lds_b128 (ASYNCcnt) overlapping the WMMA compute
//   K3: out = ao @ Wproj^T + b (fp32 out)
// Workspace: 4 x 16 MiB bf16 buffers = 64 MiB required in d_ws.
// ---------------------------------------------------------------------------

typedef __attribute__((ext_vector_type(16))) __bf16 v16bf;
typedef __attribute__((ext_vector_type(8)))  float  v8f;

#define DIMC   1024
#define NHEAD  16
#define HDIM   64
#define SEQ    2048
#define BATCH  4
#define TOKENS (BATCH * SEQ)      // 8192

union Frag16 { v16bf bf; uint4 q[2]; unsigned short us[16]; };

__device__ __forceinline__ unsigned short f2bf(float f) {
  union { float f; uint32_t u; } x; x.f = f;
  uint32_t u = x.u;
  u += 0x7FFFu + ((u >> 16) & 1u);          // round-to-nearest-even
  return (unsigned short)(u >> 16);
}

__device__ __forceinline__ v8f vzero8() {
  v8f z;
  #pragma unroll
  for (int i = 0; i < 8; ++i) z[i] = 0.0f;
  return z;
}

// async 16-byte lane copy: global (SGPR base + VGPR byte offset) -> LDS offset
__device__ __forceinline__ void async_copy16(uint32_t lds_off, uint32_t gbl_off,
                                             const void* base) {
  asm volatile("global_load_async_to_lds_b128 %0, %1, %2"
               :: "v"(lds_off), "v"(gbl_off), "s"(base) : "memory");
}
__device__ __forceinline__ void wait_async0() {
  asm volatile("s_wait_asynccnt 0x0" ::: "memory");
}

// ---------------------------------------------------------------------------
// Kernel 1: QKV GEMM.  C[m,n] = sum_k x[m,k] * wqkv[n,k]   (M=8192, N=3072, K=1024)
// 128 threads = 4 waves, block tile 64(M) x 64(N), K-step 64, LDS staged bf16.
// ---------------------------------------------------------------------------
__global__ __launch_bounds__(128) void qkv_gemm_kernel(
    const float* __restrict__ x, const float* __restrict__ wqkv,
    unsigned short* __restrict__ qb, unsigned short* __restrict__ kb,
    unsigned short* __restrict__ vtb)
{
  __shared__ __align__(16) unsigned short At[64 * 72];   // stride 72 (144B, 16B-aligned rows)
  __shared__ __align__(16) unsigned short Bt[64 * 72];

  const int nBlocks = (3 * DIMC) / 64;                   // 48
  const int mblk = blockIdx.x / nBlocks;
  const int nblk = blockIdx.x % nBlocks;
  const int m0 = mblk * 64, n0 = nblk * 64;

  const int tid  = threadIdx.x;
  const int lane = tid & 31;
  const int wv   = tid >> 5;                             // 0..3
  const int rlane = lane & 15;
  const int koff  = (lane >> 4) * 8;                     // half-wave K offset

  v8f acc[4];
  #pragma unroll
  for (int t = 0; t < 4; ++t) acc[t] = vzero8();

  for (int k0 = 0; k0 < DIMC; k0 += 64) {
    // ---- stage A (x) and B (wqkv) tiles, fp32 -> bf16 -------------------
    #pragma unroll
    for (int j = 0; j < 8; ++j) {
      int idx = tid + 128 * j;                           // 0..1023 chunks of 4
      int r   = idx >> 4;
      int c4  = (idx & 15) << 2;
      float4 a4 = *(const float4*)&x[(size_t)(m0 + r) * DIMC + k0 + c4];
      uint2 pa;
      pa.x = (uint32_t)f2bf(a4.x) | ((uint32_t)f2bf(a4.y) << 16);
      pa.y = (uint32_t)f2bf(a4.z) | ((uint32_t)f2bf(a4.w) << 16);
      *(uint2*)&At[r * 72 + c4] = pa;
      float4 b4 = *(const float4*)&wqkv[(size_t)(n0 + r) * DIMC + k0 + c4];
      uint2 pb;
      pb.x = (uint32_t)f2bf(b4.x) | ((uint32_t)f2bf(b4.y) << 16);
      pb.y = (uint32_t)f2bf(b4.z) | ((uint32_t)f2bf(b4.w) << 16);
      *(uint2*)&Bt[r * 72 + c4] = pb;
    }
    // prefetch next K-panel (A and B) while this step computes
    if (k0 + 64 < DIMC) {
      const int pr = tid & 63, ph = (tid >> 6) * 32;     // 2 threads/row, 128B apart
      __builtin_prefetch(&x[(size_t)(m0 + pr) * DIMC + k0 + 64 + ph], 0, 1);
      __builtin_prefetch(&wqkv[(size_t)(n0 + pr) * DIMC + k0 + 64 + ph], 0, 1);
    }
    __syncthreads();

    // ---- compute: wave wv owns rows [wv*16, wv*16+16), all 64 cols ------
    const int rowA = wv * 16 + rlane;
    #pragma unroll
    for (int ks = 0; ks < 2; ++ks) {
      Frag16 a;
      a.q[0] = *(const uint4*)&At[rowA * 72 + ks * 32 + koff];
      a.q[1] = *(const uint4*)&At[rowA * 72 + ks * 32 + 16 + koff];
      #pragma unroll
      for (int nt = 0; nt < 4; ++nt) {
        const int rowB = nt * 16 + rlane;
        Frag16 b;
        b.q[0] = *(const uint4*)&Bt[rowB * 72 + ks * 32 + koff];
        b.q[1] = *(const uint4*)&Bt[rowB * 72 + ks * 32 + 16 + koff];
        acc[nt] = __builtin_amdgcn_wmma_f32_16x16x32_bf16(
            false, a.bf, false, b.bf, (short)0, acc[nt], false, false);
      }
    }
    __syncthreads();
  }

  // ---- writeout: route each 16x16 tile to q / k / v^T --------------------
  const int mbase = m0 + wv * 16 + ((lane & 16) ? 8 : 0);  // C layout row base
  #pragma unroll
  for (int nt = 0; nt < 4; ++nt) {
    const int n     = n0 + nt * 16 + rlane;
    const int which = n >> 10;          // 0=q 1=k 2=v
    const int h     = (n >> 6) & 15;
    const int d     = n & 63;
    if (which == 2) {
      // v^T[b,h,d,t]: lane writes 8 contiguous tokens -> one b128 store
      const int b = mbase >> 11, t = mbase & 2047;
      uint4 pk;
      pk.x = (uint32_t)f2bf(acc[nt][0]) | ((uint32_t)f2bf(acc[nt][1]) << 16);
      pk.y = (uint32_t)f2bf(acc[nt][2]) | ((uint32_t)f2bf(acc[nt][3]) << 16);
      pk.z = (uint32_t)f2bf(acc[nt][4]) | ((uint32_t)f2bf(acc[nt][5]) << 16);
      pk.w = (uint32_t)f2bf(acc[nt][6]) | ((uint32_t)f2bf(acc[nt][7]) << 16);
      size_t base = (((size_t)(b * NHEAD + h)) * HDIM + d) * SEQ + t;
      *(uint4*)&vtb[base] = pk;
    } else {
      const float scale = (which == 0) ? 0.125f : 1.0f;   // D^-0.5 folded into q
      unsigned short* dst = (which == 0) ? qb : kb;
      #pragma unroll
      for (int i = 0; i < 8; ++i) {
        const int m = mbase + i;
        const int b = m >> 11, t = m & 2047;
        dst[(((size_t)(b * NHEAD + h)) * SEQ + t) * HDIM + d] = f2bf(acc[nt][i] * scale);
      }
    }
  }
}

// ---------------------------------------------------------------------------
// Kernel 2: flash attention.  256 threads = 8 waves share one (b,h) and 128
// consecutive queries.  K/V tiles double-buffered in LDS via async-to-LDS
// loads: tile i+1 streams in while tile i is consumed by WMMAs.
// ---------------------------------------------------------------------------
__global__ __launch_bounds__(256) void attn_kernel(
    const unsigned short* __restrict__ qb,
    const unsigned short* __restrict__ kb,
    const unsigned short* __restrict__ vtb,
    unsigned short* __restrict__ ao)
{
  __shared__ __align__(16) unsigned short Kt[2][32 * 72];   // 32 keys x 64 d (pad 72)
  __shared__ __align__(16) unsigned short Vt[2][64 * 48];   // 64 d x 32 keys (pad 48)
  __shared__ __align__(16) unsigned short Pst[8][16 * 48];  // per-wave P relayout

  const int tid  = threadIdx.x;
  const int lane = tid & 31;
  const int wv   = tid >> 5;
  const int bh   = blockIdx.x >> 4;           // (b*16+h), 0..63
  const int q0   = (blockIdx.x & 15) * 128;   // block query base
  const int b    = bh >> 4;
  const int h    = bh & 15;
  const int qt0  = q0 + wv * 16;              // this wave's 16 queries

  const unsigned short* qp = qb  + ((size_t)bh * SEQ + qt0) * HDIM;
  const unsigned short* kp = kb  +  (size_t)bh * SEQ * HDIM;
  const unsigned short* vp = vtb +  (size_t)bh * HDIM * SEQ;

  const int rlane = lane & 15;
  const int koff  = (lane >> 4) * 8;

  // Per-thread async staging chunks (16B per lane per buffer):
  //   K tile: 256 chunks -> r = tid/8 (key row), c = (tid%8)*8 (d col)
  //   V tile: 256 chunks -> vr = tid/4 (d row),  vc = (tid%4)*8 (key col)
  const int kr  = tid >> 3, kc = (tid & 7) << 3;
  const int vr  = tid >> 2, vc = (tid & 3) << 3;

  // Resident Q A-fragments (d-slices 0..31 and 32..63)
  Frag16 qa[2];
  #pragma unroll
  for (int s = 0; s < 2; ++s) {
    qa[s].q[0] = *(const uint4*)&qp[rlane * HDIM + s * 32 + koff];
    qa[s].q[1] = *(const uint4*)&qp[rlane * HDIM + s * 32 + 16 + koff];
  }

  v8f o[4];
  #pragma unroll
  for (int t = 0; t < 4; ++t) o[t] = vzero8();
  float mrow[8], lrow[8];
  #pragma unroll
  for (int i = 0; i < 8; ++i) { mrow[i] = -1e30f; lrow[i] = 0.0f; }

  unsigned short* pl = &Pst[wv][0];
  const int prow = (lane & 16) ? 8 : 0;

  // ---- prologue: async-stage tile 0 into buffer 0 ------------------------
  {
    uint32_t lk = (uint32_t)(uintptr_t)&Kt[0][kr * 72 + kc];
    async_copy16(lk, (uint32_t)((kr * HDIM + kc) * 2), kp);
    uint32_t lv = (uint32_t)(uintptr_t)&Vt[0][vr * 48 + vc];
    async_copy16(lv, (uint32_t)((vr * SEQ + vc) * 2), vp);
  }

  int it = 0;
  for (int key0 = 0; key0 < SEQ; key0 += 32, ++it) {
    const int buf = it & 1;
    wait_async0();          // this wave's chunks of tile `it` have landed
    __syncthreads();        // all waves' chunks have landed

    // kick off async staging of the NEXT tile into the other buffer
    if (key0 + 32 < SEQ) {
      uint32_t lk = (uint32_t)(uintptr_t)&Kt[buf ^ 1][kr * 72 + kc];
      async_copy16(lk, (uint32_t)(((key0 + 32 + kr) * HDIM + kc) * 2), kp);
      uint32_t lv = (uint32_t)(uintptr_t)&Vt[buf ^ 1][vr * 48 + vc];
      async_copy16(lv, (uint32_t)((vr * SEQ + key0 + 32 + vc) * 2), vp);
    }

    const unsigned short* Kb = &Kt[buf][0];
    const unsigned short* Vb = &Vt[buf][0];

    // ---- S = (q*scale) @ K^T for 32 keys: two 16x16 tiles ---------------
    v8f s0 = vzero8(), s1 = vzero8();
    #pragma unroll
    for (int s = 0; s < 2; ++s) {
      Frag16 k0f, k1f;
      k0f.q[0] = *(const uint4*)&Kb[rlane * 72 + s * 32 + koff];
      k0f.q[1] = *(const uint4*)&Kb[rlane * 72 + s * 32 + 16 + koff];
      s0 = __builtin_amdgcn_wmma_f32_16x16x32_bf16(
          false, qa[s].bf, false, k0f.bf, (short)0, s0, false, false);
      k1f.q[0] = *(const uint4*)&Kb[(16 + rlane) * 72 + s * 32 + koff];
      k1f.q[1] = *(const uint4*)&Kb[(16 + rlane) * 72 + s * 32 + 16 + koff];
      s1 = __builtin_amdgcn_wmma_f32_16x16x32_bf16(
          false, qa[s].bf, false, k1f.bf, (short)0, s1, false, false);
    }

    // ---- online softmax (row reductions across each 16-lane half) ------
    #pragma unroll
    for (int i = 0; i < 8; ++i) {
      float t = fmaxf(s0[i], s1[i]);
      t = fmaxf(t, __shfl_xor(t, 1, 32));
      t = fmaxf(t, __shfl_xor(t, 2, 32));
      t = fmaxf(t, __shfl_xor(t, 4, 32));
      t = fmaxf(t, __shfl_xor(t, 8, 32));
      const float mn    = fmaxf(mrow[i], t);
      const float alpha = __expf(mrow[i] - mn);
      mrow[i] = mn;
      const float p0 = __expf(s0[i] - mn);
      const float p1 = __expf(s1[i] - mn);
      s0[i] = p0; s1[i] = p1;
      float rs = p0 + p1;
      rs += __shfl_xor(rs, 1, 32);
      rs += __shfl_xor(rs, 2, 32);
      rs += __shfl_xor(rs, 4, 32);
      rs += __shfl_xor(rs, 8, 32);
      lrow[i] = lrow[i] * alpha + rs;
      #pragma unroll
      for (int dt = 0; dt < 4; ++dt) o[dt][i] *= alpha;
    }

    // ---- P: C-layout -> A-layout via per-wave LDS slab (bf16) -----------
    #pragma unroll
    for (int i = 0; i < 8; ++i) {
      pl[(prow + i) * 48 + rlane]      = f2bf(s0[i]);
      pl[(prow + i) * 48 + 16 + rlane] = f2bf(s1[i]);
    }
    Frag16 pa;
    pa.q[0] = *(const uint4*)&pl[rlane * 48 + koff];
    pa.q[1] = *(const uint4*)&pl[rlane * 48 + 16 + koff];

    // ---- O += P @ V (V^T in LDS -> contiguous B-frag rows) --------------
    #pragma unroll
    for (int dt = 0; dt < 4; ++dt) {
      Frag16 vb;
      vb.q[0] = *(const uint4*)&Vb[(dt * 16 + rlane) * 48 + koff];
      vb.q[1] = *(const uint4*)&Vb[(dt * 16 + rlane) * 48 + 16 + koff];
      o[dt] = __builtin_amdgcn_wmma_f32_16x16x32_bf16(
          false, pa.bf, false, vb.bf, (short)0, o[dt], false, false);
    }
    __syncthreads();   // all reads of buf done before it is refilled at it+2
  }

  // ---- finalize & store ao[b, t, h*64 + d] (bf16) ------------------------
  const int t0 = qt0 + prow;
  unsigned short* aop = ao + ((size_t)b * SEQ) * DIMC + h * HDIM;
  #pragma unroll
  for (int i = 0; i < 8; ++i) {
    const float inv = 1.0f / lrow[i];
    const size_t trow = (size_t)(t0 + i) * DIMC;
    #pragma unroll
    for (int dt = 0; dt < 4; ++dt)
      aop[trow + dt * 16 + rlane] = f2bf(o[dt][i] * inv);
  }
}

// ---------------------------------------------------------------------------
// Kernel 3: projection GEMM.  out[m,n] = sum_k ao[m,k] * wproj[n,k] + b[n]
// ---------------------------------------------------------------------------
__global__ __launch_bounds__(128) void proj_gemm_kernel(
    const unsigned short* __restrict__ ao, const float* __restrict__ wproj,
    const float* __restrict__ bias, float* __restrict__ out)
{
  __shared__ __align__(16) unsigned short At[64 * 72];
  __shared__ __align__(16) unsigned short Bt[64 * 72];

  const int nBlocks = DIMC / 64;                         // 16
  const int mblk = blockIdx.x / nBlocks;
  const int nblk = blockIdx.x % nBlocks;
  const int m0 = mblk * 64, n0 = nblk * 64;

  const int tid  = threadIdx.x;
  const int lane = tid & 31;
  const int wv   = tid >> 5;
  const int rlane = lane & 15;
  const int koff  = (lane >> 4) * 8;

  v8f acc[4];
  #pragma unroll
  for (int t = 0; t < 4; ++t) acc[t] = vzero8();

  for (int k0 = 0; k0 < DIMC; k0 += 64) {
    #pragma unroll
    for (int j = 0; j < 8; ++j) {
      int idx = tid + 128 * j;
      int r   = idx >> 4;
      int c4  = (idx & 15) << 2;
      // A already bf16: straight 8-byte copy
      *(uint2*)&At[r * 72 + c4] = *(const uint2*)&ao[(size_t)(m0 + r) * DIMC + k0 + c4];
      // B from fp32 weights
      float4 b4 = *(const float4*)&wproj[(size_t)(n0 + r) * DIMC + k0 + c4];
      uint2 pb;
      pb.x = (uint32_t)f2bf(b4.x) | ((uint32_t)f2bf(b4.y) << 16);
      pb.y = (uint32_t)f2bf(b4.z) | ((uint32_t)f2bf(b4.w) << 16);
      *(uint2*)&Bt[r * 72 + c4] = pb;
    }
    if (k0 + 64 < DIMC) {
      const int pr = tid & 63, ph = (tid >> 6) * 32;
      if (tid < 64)
        __builtin_prefetch(&ao[(size_t)(m0 + pr) * DIMC + k0 + 64], 0, 1);
      __builtin_prefetch(&wproj[(size_t)(n0 + pr) * DIMC + k0 + 64 + ph], 0, 1);
    }
    __syncthreads();

    const int rowA = wv * 16 + rlane;
    #pragma unroll
    for (int ks = 0; ks < 2; ++ks) {
      Frag16 a;
      a.q[0] = *(const uint4*)&At[rowA * 72 + ks * 32 + koff];
      a.q[1] = *(const uint4*)&At[rowA * 72 + ks * 32 + 16 + koff];
      #pragma unroll
      for (int nt = 0; nt < 4; ++nt) {
        const int rowB = nt * 16 + rlane;
        Frag16 b;
        b.q[0] = *(const uint4*)&Bt[rowB * 72 + ks * 32 + koff];
        b.q[1] = *(const uint4*)&Bt[rowB * 72 + ks * 32 + 16 + koff];
        acc[nt] = __builtin_amdgcn_wmma_f32_16x16x32_bf16(
            false, a.bf, false, b.bf, (short)0, acc[nt], false, false);
      }
    }
    __syncthreads();
  }

  const int mbase = m0 + wv * 16 + ((lane & 16) ? 8 : 0);
  #pragma unroll
  for (int nt = 0; nt < 4; ++nt) {
    const int n = n0 + nt * 16 + rlane;
    const float bv = bias[n];
    #pragma unroll
    for (int i = 0; i < 8; ++i)
      out[(size_t)(mbase + i) * DIMC + n] = acc[nt][i] + bv;
  }
}

// ---------------------------------------------------------------------------
extern "C" void kernel_launch(void* const* d_in, const int* in_sizes, int n_in,
                              void* d_out, int out_size, void* d_ws, size_t ws_size,
                              hipStream_t stream) {
  (void)in_sizes; (void)n_in; (void)out_size; (void)ws_size;
  const float* x     = (const float*)d_in[0];
  const float* wqkv  = (const float*)d_in[1];
  const float* wproj = (const float*)d_in[2];
  const float* bproj = (const float*)d_in[3];
  float* out = (float*)d_out;

  // Workspace: 4 bf16 buffers of B*H*N*D = 8,388,608 elements each (16 MiB): 64 MiB total.
  const size_t NELEM = (size_t)BATCH * NHEAD * SEQ * HDIM;
  unsigned short* qb  = (unsigned short*)d_ws;
  unsigned short* kb  = qb  + NELEM;
  unsigned short* vtb = kb  + NELEM;
  unsigned short* aob = vtb + NELEM;

  qkv_gemm_kernel<<<dim3((TOKENS / 64) * ((3 * DIMC) / 64)), 128, 0, stream>>>(
      x, wqkv, qb, kb, vtb);
  attn_kernel<<<dim3((BATCH * NHEAD) * (SEQ / 128)), 256, 0, stream>>>(
      qb, kb, vtb, aob);
  proj_gemm_kernel<<<dim3((TOKENS / 64) * (DIMC / 64)), 128, 0, stream>>>(
      aob, wproj, bproj, out);
}
